// VariationalEncoder_22548578304061
// MI455X (gfx1250) — matrix operand order repaired and checked by reading, not compiled, not run on hardware
//
#include <hip/hip_runtime.h>
#include <hip/hip_bf16.h>

typedef __attribute__((ext_vector_type(16))) _Float16 v16h;
typedef __attribute__((ext_vector_type(8)))  float    v8f;
typedef unsigned int u32x4 __attribute__((ext_vector_type(4)));
typedef int          i32x4 __attribute__((ext_vector_type(4)));
typedef int          i32x8 __attribute__((ext_vector_type(8)));

// ---------------------------------------------------------------------------
// Degree / normalization
// ---------------------------------------------------------------------------
__global__ void deg_init_kernel(float* __restrict__ deg, int n) {
    int i = blockIdx.x * blockDim.x + threadIdx.x;
    if (i < n) deg[i] = 1.0f;                      // self-loop contribution
}

__global__ void deg_accum_kernel(const int* __restrict__ dst, float* deg, int E) {
    int e = blockIdx.x * blockDim.x + threadIdx.x;
    if (e < E) atomicAdd(&deg[dst[e]], 1.0f);
}

__global__ void dinv_kernel(float* __restrict__ deg, int n) {
    int i = blockIdx.x * blockDim.x + threadIdx.x;
    if (i < n) {
        float d = deg[i];
        deg[i] = (d > 0.0f) ? rsqrtf(d) : 0.0f;
    }
}

// ---------------------------------------------------------------------------
// Weight conversion to f16
// ---------------------------------------------------------------------------
__global__ void cvt_h_kernel(const float* __restrict__ in, _Float16* __restrict__ out, int n) {
    int i = blockIdx.x * blockDim.x + threadIdx.x;
    if (i < n) out[i] = (_Float16)in[i];
}

// Wcat[k, c] = c < 64 ? W_mu[k, c] : W_ls[k, c-64];   shape [128, 128] (f16 row-major)
__global__ void build_wcat_kernel(const float* __restrict__ Wmu,
                                  const float* __restrict__ Wls,
                                  _Float16* __restrict__ out) {
    int i = blockIdx.x * blockDim.x + threadIdx.x;
    if (i < 128 * 128) {
        int k = i >> 7, c = i & 127;
        float v = (c < 64) ? Wmu[k * 64 + c] : Wls[k * 64 + (c - 64)];
        out[i] = (_Float16)v;
    }
}

// ---------------------------------------------------------------------------
// Pre-swizzle B [K,128] f16 row-major into WMMA B-fragment order:
//   out[(((kt*8 + nt)*32 + lane)*16 + e] = B[kt*32 + (e>>3)*16 + (lane>>4)*8 + (e&7), nt*16 + (lane&15)]
// Each lane of a wave then reads its 16-half fragment as one contiguous 32 B chunk.
// ---------------------------------------------------------------------------
__global__ void swizzle_b_kernel(const _Float16* __restrict__ in, _Float16* __restrict__ out,
                                 int total /* = ktiles*4096 */) {
    int i = blockIdx.x * blockDim.x + threadIdx.x;
    if (i < total) {
        int e    = i & 15;
        int lane = (i >> 4) & 31;
        int nt   = (i >> 9) & 7;
        int kt   = i >> 12;
        int k = kt * 32 + (e >> 3) * 16 + (lane >> 4) * 8 + (e & 7);
        int n = nt * 16 + (lane & 15);
        out[i] = in[k * 128 + n];
    }
}

// ---------------------------------------------------------------------------
// WMMA GEMM: C[M,128] = (RELU? max(A,0) : A)[M,KTOT] @ B[KTOT,128]
// Block = 256 threads = 8 waves; each wave owns a 16-row strip (128-row M-tile).
// B is staged per 128-deep K-chunk into LDS by the Tensor Data Mover (TDM),
// synchronized with s_wait_tensorcnt, then fragments are contiguous LDS reads.
// Fragment layouts per CDNA5 ISA 7.12.2 (f16 A 16x32, B 32x16, f32 C 16x16).
// ---------------------------------------------------------------------------
template <int KTOT, bool RELU>
__global__ __launch_bounds__(256) void gemm_wmma_kernel(
    const float* __restrict__ A, const _Float16* __restrict__ Bsw,
    float* __restrict__ C, int M) {

    constexpr int NT  = 8;         // 8 N-tiles of 16 (N = 128)
    constexpr int KCH = 128;       // K-chunk staged in LDS per TDM transfer
    __shared__ __align__(32) _Float16 Bs[KCH * 128];   // 32 KB

    const int wave = threadIdx.x >> 5;
    const int lane = threadIdx.x & 31;
    const int r    = lane & 15;
    const int kh   = lane >> 4;
    const int rowTile = blockIdx.x * 128 + wave * 16;

    v8f acc[NT] = {};

    const int  rowA = rowTile + r;
    const bool inb  = rowA < M;
    const float* Arow = A + (size_t)(inb ? rowA : 0) * KTOT;

    for (int kc = 0; kc < KTOT; kc += KCH) {
        // --- Stage 32 KB of swizzled B into LDS via TDM (one wave issues) ---
        if (wave == 0) {
            unsigned long long ga = (unsigned long long)(uintptr_t)(Bsw + (size_t)kc * 128);
            unsigned int       la = (unsigned int)(uintptr_t)(&Bs[0]);
            // D# group0: count=1 | lds_addr | global_addr(57b) | type=2
            u32x4 g0 = { 1u, la, (unsigned int)ga,
                         (unsigned int)((ga >> 32) & 0x1FFFFFFu) | (2u << 30) };
            // D# group1: data_size=4B; tensor 64 dwords x 128 rows, stride 64;
            //            tile 64 x 128 (covers the whole contiguous chunk)
            i32x8 g1 = { (int)(2u << 16),       // data_size = 2 (4 bytes)
                         (int)(64u << 16),      // tensor_dim0[15:0] = 64
                         (int)(128u << 16),     // tensor_dim1[15:0] = 128
                         (int)(64u << 16),      // tile_dim0 = 64
                         (int)128,              // tile_dim1 = 128
                         (int)64,               // tensor_dim0_stride = 64
                         0, 0 };
            i32x4 g2 = { 0, 0, 0, 0 };          // no dims 2..4 (2-D tensor)
            i32x4 g3 = { 0, 0, 0, 0 };
            i32x8 gx = { 0, 0, 0, 0, 0, 0, 0, 0 };  // extra arg of 6-arg builtin (zero-filled)
            __builtin_amdgcn_tensor_load_to_lds(g0, g1, g2, g3, gx, 0);
            __builtin_amdgcn_s_wait_tensorcnt(0);
        }
        __syncthreads();

        // --- Compute 4 K-tiles (4 x K=32) out of this chunk ---
#pragma unroll
        for (int kt = 0; kt < KCH / 32; ++kt) {
            const int k0 = kc + kt * 32;
            __builtin_prefetch(Arow + k0 + 128, 0, 1);   // global_prefetch_b8

            v16h a = {};
            if (inb) {
#pragma unroll
                for (int e = 0; e < 8; ++e) {
                    float f0 = Arow[k0 + kh * 8 + e];
                    float f1 = Arow[k0 + 16 + kh * 8 + e];
                    if (RELU) { f0 = fmaxf(f0, 0.0f); f1 = fmaxf(f1, 0.0f); }
                    a[e]     = (_Float16)f0;
                    a[e + 8] = (_Float16)f1;
                }
            }

#pragma unroll
            for (int nt = 0; nt < NT; ++nt) {
                const v16h b = *(const v16h*)&Bs[(((kt * NT) + nt) * 32 + lane) * 16];
                // (neg_a, A, neg_b, B, c_mod, C, reuse_a, reuse_b)
                acc[nt] = __builtin_amdgcn_wmma_f32_16x16x32_f16(
                    false, a, false, b, (short)0, acc[nt], false, false);
            }
        }
        __syncthreads();   // chunk fully consumed before next TDM overwrite
    }

#pragma unroll
    for (int nt = 0; nt < NT; ++nt) {
#pragma unroll
        for (int j = 0; j < 8; ++j) {
            int row = rowTile + kh * 8 + j;
            if (row < M) C[(size_t)row * 128 + nt * 16 + r] = acc[nt][j];
        }
    }
}

// ---------------------------------------------------------------------------
// Aggregation, layer 1 (128 channels):
//   out[n,c] = b1[c] + H[n,c]*dinv[n]^2 + sum_{e: dst=n} H[src,c]*dinv[src]*dinv[n]
// ---------------------------------------------------------------------------
__global__ void agg_init_kernel(const float* __restrict__ H, const float* __restrict__ bias,
                                const float* __restrict__ dinv, float* __restrict__ out, int n) {
    long long i = (long long)blockIdx.x * blockDim.x + threadIdx.x;
    if (i < ((long long)n << 7)) {
        int node = (int)(i >> 7), c = (int)(i & 127);
        float di = dinv[node];
        out[i] = bias[c] + H[i] * di * di;
    }
}

__global__ void agg_edge_kernel(const float* __restrict__ H, const int* __restrict__ src,
                                const int* __restrict__ dst, const float* __restrict__ dinv,
                                float* out, int E) {
    long long t = (long long)blockIdx.x * blockDim.x + threadIdx.x;
    if (t < ((long long)E << 7)) {
        int e = (int)(t >> 7), c = (int)(t & 127);
        int s = src[e], d = dst[e];
        atomicAdd(&out[(size_t)d * 128 + c], H[(size_t)s * 128 + c] * dinv[s] * dinv[d]);
    }
}

// ---------------------------------------------------------------------------
// Aggregation, layers 2/3 fused (H2 = [mu_pre | ls_pre], 64+64 channels)
// ---------------------------------------------------------------------------
__global__ void agg_init_split_kernel(const float* __restrict__ H2,
                                      const float* __restrict__ bmu, const float* __restrict__ bls,
                                      const float* __restrict__ dinv,
                                      float* __restrict__ mu, float* __restrict__ ls, int n) {
    long long i = (long long)blockIdx.x * blockDim.x + threadIdx.x;
    if (i < ((long long)n << 7)) {
        int node = (int)(i >> 7), c = (int)(i & 127);
        float di = dinv[node];
        float v = H2[i] * di * di;
        if (c < 64) mu[(size_t)node * 64 + c]        = bmu[c]      + v;
        else        ls[(size_t)node * 64 + (c - 64)] = bls[c - 64] + v;
    }
}

__global__ void agg_edge_split_kernel(const float* __restrict__ H2, const int* __restrict__ src,
                                      const int* __restrict__ dst, const float* __restrict__ dinv,
                                      float* mu, float* ls, int E) {
    long long t = (long long)blockIdx.x * blockDim.x + threadIdx.x;
    if (t < ((long long)E << 7)) {
        int e = (int)(t >> 7), c = (int)(t & 127);
        int s = src[e], d = dst[e];
        float v = H2[(size_t)s * 128 + c] * dinv[s] * dinv[d];
        if (c < 64) atomicAdd(&mu[(size_t)d * 64 + c], v);
        else        atomicAdd(&ls[(size_t)d * 64 + (c - 64)], v);
    }
}

// ---------------------------------------------------------------------------
// Host launcher
// ---------------------------------------------------------------------------
extern "C" void kernel_launch(void* const* d_in, const int* in_sizes, int n_in,
                              void* d_out, int out_size, void* d_ws, size_t ws_size,
                              hipStream_t stream) {
    (void)n_in; (void)out_size; (void)ws_size;

    const float* x   = (const float*)d_in[0];
    const int*   ei  = (const int*)d_in[1];
    const float* W1  = (const float*)d_in[2];
    const float* b1  = (const float*)d_in[3];
    const float* Wmu = (const float*)d_in[4];
    const float* bmu = (const float*)d_in[5];
    const float* Wls = (const float*)d_in[6];
    const float* bls = (const float*)d_in[7];

    const int IN = 512, HID = 128, OUT = 64;
    const int M = in_sizes[0] / IN;     // 100000 nodes
    const int E = in_sizes[1] / 2;      // 1600000 edges
    const int* src = ei;
    const int* dst = ei + E;

    // Workspace carve-up
    char*  ws  = (char*)d_ws;
    size_t off = 0;
    auto carve = [&](size_t bytes) -> void* {
        void* p = ws + off;
        off += (bytes + 255) & ~(size_t)255;
        return p;
    };
    float*    dinv = (float*)   carve((size_t)M * sizeof(float));
    float*    H0   = (float*)   carve((size_t)M * HID * sizeof(float));   // x @ W1
    float*    H1   = (float*)   carve((size_t)M * HID * sizeof(float));   // aggregated (pre-relu)
    float*    H2   = (float*)   carve((size_t)M * HID * sizeof(float));   // relu(H1) @ [Wmu|Wls]
    _Float16* W1h  = (_Float16*)carve((size_t)IN * HID * sizeof(_Float16));   // row-major f16
    _Float16* Wch  = (_Float16*)carve((size_t)HID * HID * sizeof(_Float16));  // row-major f16
    _Float16* W1sw = (_Float16*)carve((size_t)IN * HID * sizeof(_Float16));   // fragment-swizzled
    _Float16* Wcsw = (_Float16*)carve((size_t)HID * HID * sizeof(_Float16));  // fragment-swizzled

    float* mu = (float*)d_out;
    float* ls = mu + (size_t)M * OUT;

    const int T = 256;
    // 1) symmetric normalization dinv = (deg with self-loop)^(-1/2)
    deg_init_kernel <<<(M + T - 1) / T, T, 0, stream>>>(dinv, M);
    deg_accum_kernel<<<(E + T - 1) / T, T, 0, stream>>>(dst, dinv, E);
    dinv_kernel     <<<(M + T - 1) / T, T, 0, stream>>>(dinv, M);

    // 2) weights -> f16, then swizzle into WMMA fragment order
    cvt_h_kernel     <<<(IN * HID + T - 1) / T, T, 0, stream>>>(W1, W1h, IN * HID);
    build_wcat_kernel<<<(HID * HID + T - 1) / T, T, 0, stream>>>(Wmu, Wls, Wch);
    swizzle_b_kernel <<<(IN * HID + T - 1) / T, T, 0, stream>>>(W1h, W1sw, IN * HID);
    swizzle_b_kernel <<<(HID * HID + T - 1) / T, T, 0, stream>>>(Wch, Wcsw, HID * HID);

    // 3) layer-1 GEMM (WMMA + TDM-staged B), then GCN aggregate + bias
    gemm_wmma_kernel<512, false><<<(M + 127) / 128, 256, 0, stream>>>(x, W1sw, H0, M);
    long long nc = (long long)M << 7;           // M*128 element-threads
    long long ec = (long long)E << 7;           // E*128 element-threads
    agg_init_kernel <<<(int)((nc + T - 1) / T), T, 0, stream>>>(H0, b1, dinv, H1, M);
    agg_edge_kernel <<<(int)((ec + T - 1) / T), T, 0, stream>>>(H0, src, dst, dinv, H1, E);

    // 4) layers 2+3 fused GEMM (relu(A) on the fly), split aggregation -> mu / logstd
    gemm_wmma_kernel<128, true><<<(M + 127) / 128, 256, 0, stream>>>(H1, Wcsw, H2, M);
    agg_init_split_kernel<<<(int)((nc + T - 1) / T), T, 0, stream>>>(H2, bmu, bls, dinv, mu, ls, M);
    agg_edge_split_kernel<<<(int)((ec + T - 1) / T), T, 0, stream>>>(H2, src, dst, dinv, mu, ls, E);
}
